// MSAttention_45183055954448
// MI455X (gfx1250) — matrix operand order repaired and checked by reading, not compiled, not run on hardware
//
#include <hip/hip_runtime.h>
#include <cstdint>

typedef __attribute__((ext_vector_type(16))) _Float16 v16h;
typedef __attribute__((ext_vector_type(8)))  float    v8f;
typedef __attribute__((ext_vector_type(4)))  float    v4f;

#define DIM 512
#define HEADS 8
#define HEAD_DIM 64
#define HIDDEN 2048
#define SCALE 0.125f
#define SP 260   // padded score row stride (floats); 1040B rows -> 16B aligned

// GEMM LDS geometry (f32 staging, double-buffered)
#define AS_PAD 36            // 144B rows: 16B aligned, conflict-free
#define BS_PAD 36            // 144B rows: 16B aligned (b128 operand reads)
#define AS_FLOATS (128 * AS_PAD)             // 4608
#define BS_FLOATS (128 * BS_PAD)             // 4608
#define BUF_FLOATS (AS_FLOATS + BS_FLOATS)   // 9216
#define GEMM_LDS_BYTES (2 * BUF_FLOATS * 4)  // 73728

// ---- CDNA5 16-bit WMMA operand element maps (wave32) ----
// A 16x32 (MxK): lane m=l&15, half kh=l>>4; elements are two contiguous runs:
//   h[0..7]  <- K = kh*8 .. kh*8+7        h[8..15] <- K = 16+kh*8 .. 16+kh*8+7
// B 32x16 (KxN): lane n=l&15, half kh=l>>4; same two runs as A:
//   h[0..7]  <- K = kh*8 .. kh*8+7        h[8..15] <- K = 16+kh*8 .. 16+kh*8+7

union H16 { v16h v; _Float16 h[16]; unsigned int u[8]; };

__device__ __forceinline__ void packH16(H16& o, v4f a0, v4f a1, v4f b0, v4f b1) {
    #pragma unroll
    for (int i = 0; i < 4; ++i) {
        o.h[i]      = (_Float16)a0[i];
        o.h[4 + i]  = (_Float16)a1[i];
        o.h[8 + i]  = (_Float16)b0[i];
        o.h[12 + i] = (_Float16)b1[i];
    }
}
// A-pattern operand from a row pointer (two 8-float runs)
__device__ __forceinline__ void loadA16(H16& o, const float* row, int kh) {
    const v4f* g0 = (const v4f*)(row + kh * 8);
    const v4f* g1 = (const v4f*)(row + 16 + kh * 8);
    packH16(o, g0[0], g0[1], g1[0], g1[1]);
}
// B-pattern operand from a row pointer (two 8-float runs, same as A)
__device__ __forceinline__ void loadB16(H16& o, const float* row, int kh) {
    const v4f* g0 = (const v4f*)(row + kh * 8);
    const v4f* g1 = (const v4f*)(row + 16 + kh * 8);
    packH16(o, g0[0], g0[1], g1[0], g1[1]);
}

// ---------------------------------------------------------------------------
// Bilinear resize 32x32 -> 64x64 (token-major), jax coord = (i+0.5)/2 - 0.5
// ---------------------------------------------------------------------------
__global__ void resize_up_kernel(const float* __restrict__ y, float* __restrict__ out) {
    size_t idx = (size_t)blockIdx.x * 256 + threadIdx.x;   // 8*4096*512
    int c = (int)(idx & 511);
    int n = (int)((idx >> 9) & 4095);
    int b = (int)(idx >> 21);
    int i = n >> 6, j = n & 63;

    int ti = i >> 1, tj = j >> 1;
    int lh, hh, lw, hw; float w0h, w1h, w0w, w1w;
    if ((i & 1) == 0) { lh = ti - 1; hh = ti;     w0h = 0.25f; w1h = 0.75f; }
    else              { lh = ti;     hh = ti + 1; w0h = 0.75f; w1h = 0.25f; }
    if ((j & 1) == 0) { lw = tj - 1; hw = tj;     w0w = 0.25f; w1w = 0.75f; }
    else              { lw = tj;     hw = tj + 1; w0w = 0.75f; w1w = 0.25f; }
    lh = lh < 0 ? 0 : lh; hh = hh > 31 ? 31 : hh;
    lw = lw < 0 ? 0 : lw; hw = hw > 31 ? 31 : hw;

    const float* yb = y + (size_t)b * 1024 * 512;
    float v = w0h * (w0w * yb[(size_t)(lh * 32 + lw) * 512 + c] +
                     w1w * yb[(size_t)(lh * 32 + hw) * 512 + c]) +
              w1h * (w0w * yb[(size_t)(hh * 32 + lw) * 512 + c] +
                     w1w * yb[(size_t)(hh * 32 + hw) * 512 + c]);
    out[idx] = v;
}

// 64x64 -> 32x32 (antialias=False): coord = 2i+0.5 -> plain 2x2 average
__global__ void resize_down_kernel(const float* __restrict__ x, float* __restrict__ out) {
    size_t idx = (size_t)blockIdx.x * 256 + threadIdx.x;   // 8*1024*512
    int c = (int)(idx & 511);
    int n = (int)((idx >> 9) & 1023);
    int b = (int)(idx >> 19);
    int i = n >> 5, j = n & 31;
    const float* xb = x + (size_t)b * 4096 * 512;
    int r0 = 2 * i, c0 = 2 * j;
    float v = 0.25f * (xb[(size_t)(r0 * 64 + c0) * 512 + c] +
                       xb[(size_t)(r0 * 64 + c0 + 1) * 512 + c] +
                       xb[(size_t)((r0 + 1) * 64 + c0) * 512 + c] +
                       xb[(size_t)((r0 + 1) * 64 + c0 + 1) * 512 + c]);
    out[idx] = v;
}

// ---------------------------------------------------------------------------
// Async stage of one K-step tile: A 128x32 (b128, row-major) and
// W 32x128 (b32, transposed scatter into Bs[n][k]).  20 async ops/lane/tile.
// ---------------------------------------------------------------------------
__device__ __forceinline__ void gemm_stage_async(const float* A, const float* W,
                                                 int K, int N, int tile_m, int tile_n,
                                                 int k0, uint32_t ldsbase, int tid) {
    #pragma unroll
    for (int i = 0; i < 4; ++i) {              // A tile: 1024 vec4s
        int e = tid + i * 256;
        int r = e >> 3, c4 = (e & 7) << 2;
        uint64_t ga = (uint64_t)(uintptr_t)(A + (size_t)(tile_m + r) * K + k0 + c4);
        uint32_t la = ldsbase + (uint32_t)(((r * AS_PAD) + c4) << 2);
        asm volatile("global_load_async_to_lds_b128 %0, %1, off"
                     :: "v"(la), "v"(ga) : "memory");
    }
    #pragma unroll
    for (int i = 0; i < 16; ++i) {             // W tile: 4096 scalars, transpose scatter
        int e = tid + i * 256;
        int kr = e >> 7, nc = e & 127;
        uint64_t ga = (uint64_t)(uintptr_t)(W + (size_t)(k0 + kr) * N + tile_n + nc);
        uint32_t la = ldsbase + (uint32_t)(AS_FLOATS * 4) + (uint32_t)(((nc * BS_PAD) + kr) << 2);
        asm volatile("global_load_async_to_lds_b32 %0, %1, off"
                     :: "v"(la), "v"(ga) : "memory");
    }
}

// ---------------------------------------------------------------------------
// WMMA GEMM: C = epilogue(A[M,K] @ W[K,N] + bias)
// 256 threads (8 waves); block tile 128x128; wave tile 32x64 => 8 WMMA / K-step.
// Double-buffered async f32 LDS staging (ASYNCcnt pipeline), f16 at operand build.
// mode 0: store  1: +R  2: exact GELU  3: +R and remap rows into concat output
// ---------------------------------------------------------------------------
__global__ __launch_bounds__(256)
void gemm_f16wmma_kernel(const float* __restrict__ A, const float* __restrict__ W,
                         const float* __restrict__ bias, const float* __restrict__ R,
                         float* __restrict__ Cout,
                         int M, int N, int K, int mode,
                         int g_offset, int rows_pb, int out_base, int out_stride) {
    extern __shared__ float gsm[];
    const int tid  = threadIdx.x;
    const int lane = tid & 31;
    const int wave = tid >> 5;
    const int wr = wave & 3;               // 4 row groups of 32
    const int wc = wave >> 2;              // 2 col groups of 64
    const int tile_m = blockIdx.x * 128;
    const int tile_n = blockIdx.y * 128;
    const int kh = lane >> 4, ml = lane & 15;
    const uint32_t lds0 = (uint32_t)(uintptr_t)gsm;

    v8f acc[2][4] = {};
    const int nsteps = K / 32;

    gemm_stage_async(A, W, K, N, tile_m, tile_n, 0, lds0, tid);

    for (int s = 0; s < nsteps; ++s) {
        const int sel = s & 1;
        if (s + 1 < nsteps) {
            gemm_stage_async(A, W, K, N, tile_m, tile_n, (s + 1) * 32,
                             lds0 + (uint32_t)((sel ^ 1) * BUF_FLOATS * 4), tid);
            asm volatile("s_wait_asynccnt 0x14" ::: "memory");   // <=20: current tile done
        } else {
            asm volatile("s_wait_asynccnt 0x0" ::: "memory");
        }
        __syncthreads();

        const float* Asf = gsm + (size_t)sel * BUF_FLOATS;
        const float* Bsf = Asf + AS_FLOATS;

        H16 a[2];
        #pragma unroll
        for (int h2 = 0; h2 < 2; ++h2)
            loadA16(a[h2], Asf + (size_t)(wr * 32 + h2 * 16 + ml) * AS_PAD, kh);

        #pragma unroll
        for (int dt = 0; dt < 4; ++dt) {
            H16 bm;
            loadB16(bm, Bsf + (size_t)(wc * 64 + dt * 16 + ml) * BS_PAD, kh);
            #pragma unroll
            for (int h2 = 0; h2 < 2; ++h2)
                acc[h2][dt] = __builtin_amdgcn_wmma_f32_16x16x32_f16(
                    false, a[h2].v, false, bm.v, (short)0, acc[h2][dt], false, false);
        }
        __syncthreads();   // all reads done before this buffer is re-targeted
    }

    // epilogue; C-layout: VGPR r -> (M = r + kh*8, N = ml)
    #pragma unroll
    for (int h2 = 0; h2 < 2; ++h2) {
        #pragma unroll
        for (int dt = 0; dt < 4; ++dt) {
            int n = tile_n + wc * 64 + dt * 16 + ml;
            float bv = bias[n];
            #pragma unroll
            for (int r = 0; r < 8; ++r) {
                int m = tile_m + wr * 32 + h2 * 16 + kh * 8 + r;
                float v = acc[h2][dt][r] + bv;
                if (mode == 1 || mode == 3) v += R[(size_t)m * N + n];
                if (mode == 2) v = 0.5f * v * (1.0f + erff(v * 0.70710678118654752f));
                size_t oidx;
                if (mode == 3) {
                    int g  = g_offset + m;
                    int bb = g / rows_pb;
                    int nn = g - bb * rows_pb;
                    oidx = ((size_t)bb * out_stride + out_base + nn) * (size_t)N + n;
                } else {
                    oidx = (size_t)m * N + n;
                }
                Cout[oidx] = v;
            }
        }
    }
}

// ---------------------------------------------------------------------------
// Windowed attention, one block per (batch, window, head). T=256, d=64.
// 512 threads = 16 waves; wave w owns query rows [16w,16w+16).
// Dynamic LDS: S[256][260] f32 + rowsum[256] + tokrow[256].
// ---------------------------------------------------------------------------
__global__ __launch_bounds__(512)
void attn_kernel(const float* __restrict__ Q, const float* __restrict__ Kt,
                 const float* __restrict__ V, float* __restrict__ O,
                 int side, int nW, int Npb) {
    extern __shared__ float smem[];
    float* S      = smem;                       // 256*260
    float* rowsum = smem + 256 * SP;            // 256
    int*   tokrow = (int*)(rowsum + 256);       // 256 (token row * 512)

    const int tid  = threadIdx.x;
    const int lane = tid & 31;
    const int wave = tid >> 5;
    int idx = blockIdx.x;
    const int head = idx & 7;  idx >>= 3;
    const int ww   = idx % nW; idx /= nW;
    const int wh   = idx % nW; idx /= nW;
    const int b    = idx;
    const int hb   = head * HEAD_DIM;

    if (tid < 256) {
        int i = tid >> 4, j = tid & 15;
        tokrow[tid] = (b * Npb + (wh * 16 + i) * side + (ww * 16 + j)) * DIM;
    }
    __syncthreads();

    const int m0 = wave * 16;
    const int kh = lane >> 4;
    const int ml = lane & 15;

    // phase 1: S = scale * Q K^T   (A = Q rows, B = K-token columns)
    H16 qlo, qhi;
    {
        const float* qp = Q + tokrow[m0 + ml] + hb;
        loadA16(qlo, qp, kh);
        loadA16(qhi, qp + 32, kh);
    }
    for (int n0 = 0; n0 < 256; n0 += 16) {
        H16 klo, khi;
        const float* kp = Kt + tokrow[n0 + ml] + hb;
        loadB16(klo, kp, kh);
        loadB16(khi, kp + 32, kh);
        v8f sacc = {};
        sacc = __builtin_amdgcn_wmma_f32_16x16x32_f16(false, qlo.v, false, klo.v,
                                                      (short)0, sacc, false, false);
        sacc = __builtin_amdgcn_wmma_f32_16x16x32_f16(false, qhi.v, false, khi.v,
                                                      (short)0, sacc, false, false);
        #pragma unroll
        for (int r = 0; r < 8; ++r)
            S[(m0 + kh * 8 + r) * SP + n0 + ml] = sacc[r] * SCALE;
    }
    __syncthreads();

    // phase 2: softmax (sum deferred to output scale), vectorized row scans
    if (tid < 256) {
        float* row = S + (size_t)tid * SP;
        v4f mv = *(const v4f*)row;
        for (int i = 4; i < 256; i += 4) {
            v4f t = *(const v4f*)(row + i);
            #pragma unroll
            for (int e = 0; e < 4; ++e) mv[e] = fmaxf(mv[e], t[e]);
        }
        float mx = fmaxf(fmaxf(mv[0], mv[1]), fmaxf(mv[2], mv[3]));
        float sum = 0.0f;
        for (int i = 0; i < 256; i += 4) {
            v4f t = *(const v4f*)(row + i);
            #pragma unroll
            for (int e = 0; e < 4; ++e) { t[e] = expf(t[e] - mx); sum += t[e]; }
            *(v4f*)(row + i) = t;
        }
        rowsum[tid] = sum;
    }
    __syncthreads();

    // phase 3: O = P V   (A = P rows from LDS, B = V-channel columns from global)
    v8f oacc[4] = {};
    for (int nc = 0; nc < 256; nc += 32) {
        H16 pa;
        loadA16(pa, S + (size_t)(m0 + ml) * SP + nc, kh);
        #pragma unroll
        for (int dt = 0; dt < 4; ++dt) {
            H16 vb;
            int d = hb + dt * 16 + ml;
            #pragma unroll
            for (int j = 0; j < 8; ++j) {
                int k = (j < 4) ? kh * 8 + 2 * j : 16 + kh * 8 + 2 * (j - 4);       // B-pattern: two K runs (as A)
                vb.h[2 * j]     = (_Float16)V[tokrow[nc + k] + d];
                vb.h[2 * j + 1] = (_Float16)V[tokrow[nc + k + 1] + d];
            }
            oacc[dt] = __builtin_amdgcn_wmma_f32_16x16x32_f16(
                false, pa.v, false, vb.v, (short)0, oacc[dt], false, false);
        }
    }
    #pragma unroll
    for (int r = 0; r < 8; ++r) {
        int row   = m0 + kh * 8 + r;
        float rin = 1.0f / rowsum[row];
        int obase = tokrow[row] + hb;
        #pragma unroll
        for (int dt = 0; dt < 4; ++dt)
            O[obase + dt * 16 + ml] = oacc[dt][r] * rin;
    }
}

// ---------------------------------------------------------------------------
extern "C" void kernel_launch(void* const* d_in, const int* in_sizes, int n_in,
                              void* d_out, int out_size, void* d_ws, size_t ws_size,
                              hipStream_t stream) {
    (void)in_sizes; (void)n_in; (void)out_size; (void)ws_size;
    const float* x  = (const float*)d_in[0];   // (8,4096,512)
    const float* y  = (const float*)d_in[1];   // (8,1024,512)
    const float* Wq = (const float*)d_in[2];  const float* bq = (const float*)d_in[3];
    const float* Wk = (const float*)d_in[4];  const float* bk = (const float*)d_in[5];
    const float* Wv = (const float*)d_in[6];  const float* bv = (const float*)d_in[7];
    const float* Wo = (const float*)d_in[8];  const float* bo = (const float*)d_in[9];
    const float* W1 = (const float*)d_in[10]; const float* b1 = (const float*)d_in[11];
    const float* W2 = (const float*)d_in[12]; const float* b2 = (const float*)d_in[13];
    float* out = (float*)d_out;
    float* ws  = (float*)d_ws;

    const size_t SXn = 32768ull * 512;
    const size_t SYn = 8192ull * 512;
    float* rx = ws;                       // resized y (64x64 tokens)
    float* ry = rx + SXn;                 // resized x (32x32 tokens)
    float* qx = ry + SYn;
    float* kx = qx + SXn;
    float* vx = kx + SXn;
    float* qy = vx + SXn;
    float* ky = qy + SYn;
    float* vy = ky + SYn;
    float* hb_ = vy + SYn;                // MLP hidden chunk: 2048 x 2048
    float* ax = rx;                       // attn-out x (rx dead after Kx GEMM)
    float* ay = ry;
    float* tx = kx;                       // tokens x (kx dead after attention)
    float* ty = ky;

    const size_t attn_lds = (size_t)(256 * SP + 256 + 256) * sizeof(float);
    (void)hipFuncSetAttribute((const void*)attn_kernel,
                              hipFuncAttributeMaxDynamicSharedMemorySize, (int)attn_lds);
    (void)hipFuncSetAttribute((const void*)gemm_f16wmma_kernel,
                              hipFuncAttributeMaxDynamicSharedMemorySize, GEMM_LDS_BYTES);

    resize_up_kernel<<<(8 * 4096 * 512) / 256, 256, 0, stream>>>(y, rx);
    resize_down_kernel<<<(8 * 1024 * 512) / 256, 256, 0, stream>>>(x, ry);

    dim3 blk(256);
    const size_t GL = GEMM_LDS_BYTES;
    gemm_f16wmma_kernel<<<dim3(256, 4), blk, GL, stream>>>(x,  Wq, bq, nullptr, qx, 32768, 512, 512, 0, 0, 0, 0, 0);
    gemm_f16wmma_kernel<<<dim3(256, 4), blk, GL, stream>>>(rx, Wk, bk, nullptr, kx, 32768, 512, 512, 0, 0, 0, 0, 0);
    gemm_f16wmma_kernel<<<dim3(256, 4), blk, GL, stream>>>(x,  Wv, bv, nullptr, vx, 32768, 512, 512, 0, 0, 0, 0, 0);
    gemm_f16wmma_kernel<<<dim3(64, 4),  blk, GL, stream>>>(y,  Wq, bq, nullptr, qy,  8192, 512, 512, 0, 0, 0, 0, 0);
    gemm_f16wmma_kernel<<<dim3(64, 4),  blk, GL, stream>>>(ry, Wk, bk, nullptr, ky,  8192, 512, 512, 0, 0, 0, 0, 0);
    gemm_f16wmma_kernel<<<dim3(64, 4),  blk, GL, stream>>>(y,  Wv, bv, nullptr, vy,  8192, 512, 512, 0, 0, 0, 0, 0);

    attn_kernel<<<8 * 16 * HEADS, 512, attn_lds, stream>>>(qx, kx, vx, ax, 64, 4, 4096);
    attn_kernel<<<8 *  4 * HEADS, 512, attn_lds, stream>>>(qy, ky, vy, ay, 32, 2, 1024);

    gemm_f16wmma_kernel<<<dim3(256, 4), blk, GL, stream>>>(ax, Wo, bo, x, tx, 32768, 512, 512, 1, 0, 0, 0, 0);
    gemm_f16wmma_kernel<<<dim3(64, 4),  blk, GL, stream>>>(ay, Wo, bo, y, ty,  8192, 512, 512, 1, 0, 0, 0, 0);

    for (int c = 0; c < 16; ++c) {                       // branch x MLP: 16 x 2048 rows
        const float* ta = tx + (size_t)c * 2048 * 512;
        gemm_f16wmma_kernel<<<dim3(16, 16), blk, GL, stream>>>(ta, W1, b1, nullptr, hb_,
                                                               2048, 2048, 512, 2, 0, 0, 0, 0);
        gemm_f16wmma_kernel<<<dim3(16, 4), blk, GL, stream>>>(hb_, W2, b2, ta, out,
                                                              2048, 512, 2048, 3,
                                                              c * 2048, 4096, 0, 5120);
    }
    for (int c = 0; c < 4; ++c) {                        // branch y MLP: 4 x 2048 rows
        const float* ta = ty + (size_t)c * 2048 * 512;
        gemm_f16wmma_kernel<<<dim3(16, 16), blk, GL, stream>>>(ta, W1, b1, nullptr, hb_,
                                                               2048, 2048, 512, 2, 0, 0, 0, 0);
        gemm_f16wmma_kernel<<<dim3(16, 4), blk, GL, stream>>>(hb_, W2, b2, ta, out,
                                                              2048, 512, 2048, 3,
                                                              c * 2048, 1024, 4096, 5120);
    }
}
